// CosinesimSegHead_3513283248384
// MI455X (gfx1250) — matrix operand order repaired and checked by reading, not compiled, not run on hardware
//
#include <hip/hip_runtime.h>

// ---------------------------------------------------------------------------
// CosinesimSegHead for MI455X (gfx1250, wave32, WMMA + async-to-LDS)
// B=8, HW=16384 -> M=131072 rows, C=256, E=1024
// ---------------------------------------------------------------------------

#define M_TOTAL (8 * 16384)
#define CDIM    256
#define EDIM    1024
#define M_BLK   128            // rows per workgroup (8 waves x 16 rows)
#define N_CHUNK 64             // codebook rows per LDS chunk (double buffered)
#define ROW_U   (CDIM / 2)     // row stride in packed-bf16 uints (=128)
#define N_CHUNKS (EDIM / N_CHUNK)          // 16
#define CHUNK_B128 (N_CHUNK * ROW_U / 4)   // 2048 uint4 per chunk
#define B128_PER_THREAD (CHUNK_B128 / 256) // 8

typedef __attribute__((ext_vector_type(16))) __bf16 v16bf;
typedef __attribute__((ext_vector_type(8)))  float  v8f;
typedef __attribute__((ext_vector_type(4)))  float  f4;   // clang vector (NT-capable)

union FragAB { uint4 q[2]; v16bf v; };

__device__ __forceinline__ unsigned pk_bf16(float a, float b) {
    __bf16 x = (__bf16)a, y = (__bf16)b;                  // RNE convert
    unsigned short ux = __builtin_bit_cast(unsigned short, x);
    unsigned short uy = __builtin_bit_cast(unsigned short, y);
    return (unsigned)ux | ((unsigned)uy << 16);
}

// full-wave (32 lane) sum-of-squares reduce -> 1/max(sqrt(s), eps)
__device__ __forceinline__ float row_scale(float s) {
    s += __shfl_xor(s, 16);
    s += __shfl_xor(s, 8);
    s += __shfl_xor(s, 4);
    s += __shfl_xor(s, 2);
    s += __shfl_xor(s, 1);
    float n = fmaxf(sqrtf(s), 1e-12f);
    return 1.0f / n;
}

// GLOBAL_LOAD_ASYNC_TO_LDS_B128, GV mode: LDS address comes from the vdst VGPR;
// tracked by ASYNCcnt (no VGPR staging).
__device__ __forceinline__ void async_load_b128(unsigned lds_off, const void* gaddr) {
    asm volatile("global_load_async_to_lds_b128 %0, %1, off"
                 :: "v"(lds_off), "v"(gaddr)
                 : "memory");
}

__device__ __forceinline__ void wait_async0() {
#if __has_builtin(__builtin_amdgcn_s_wait_asynccnt)
    __builtin_amdgcn_s_wait_asynccnt(0);
#else
    asm volatile("s_wait_asynccnt 0x0" ::: "memory");
#endif
}

// ---------------------------------------------------------------------------
// Kernel 1: l2-normalize codebook -> f32 copy + packed bf16 copy; zero flags.
// 128 blocks x 256 threads, one wave per codebook row.
// ---------------------------------------------------------------------------
__global__ void vq_prep_weights(const float* __restrict__ w,
                                float*       __restrict__ w_norm,
                                unsigned*    __restrict__ w_bf16,
                                unsigned*    __restrict__ flags) {
    const int tid  = threadIdx.x;
    const int wave = tid >> 5;
    const int lane = tid & 31;
    const int e    = blockIdx.x * 8 + wave;

    const f4* src = (const f4*)(w + (size_t)e * CDIM);
    f4 p = src[lane * 2 + 0];
    f4 q = src[lane * 2 + 1];
    float s = p.x*p.x + p.y*p.y + p.z*p.z + p.w*p.w
            + q.x*q.x + q.y*q.y + q.z*q.z + q.w*q.w;
    float sc = row_scale(s);
    p *= sc;
    q *= sc;

    f4* dn = (f4*)(w_norm + (size_t)e * CDIM);
    dn[lane * 2 + 0] = p;
    dn[lane * 2 + 1] = q;

    uint4 pb;
    pb.x = pk_bf16(p.x, p.y);
    pb.y = pk_bf16(p.z, p.w);
    pb.z = pk_bf16(q.x, q.y);
    pb.w = pk_bf16(q.z, q.w);
    *(uint4*)(w_bf16 + (size_t)e * ROW_U + lane * 4) = pb;

    if (blockIdx.x < 4) flags[blockIdx.x * 256 + tid] = 0u;   // EDIM flags
}

// ---------------------------------------------------------------------------
// Kernel 2: fused normalize(x) -> bf16 WMMA GEMM -> distance + argmax +
//           quantized gather. 1024 blocks x 256 threads (8 waves).
// LDS: A 64KB | B ping 32KB | B pong 32KB | idx 512B  (= 128.5 KB)
// ---------------------------------------------------------------------------
__global__ void __launch_bounds__(256)
vq_main(const float*    __restrict__ x,
        const unsigned* __restrict__ w_bf16,
        const float*    __restrict__ w_norm,
        unsigned*       __restrict__ flags,
        float*          __restrict__ quant_out,
        float*          __restrict__ dist_out,
        float*          __restrict__ embed_out) {
    extern __shared__ unsigned smem[];
    unsigned* Al  = smem;                         // M_BLK x ROW_U uints
    unsigned* Bl0 = smem + M_BLK * ROW_U;         // N_CHUNK x ROW_U uints
    unsigned* Bl1 = Bl0 + N_CHUNK * ROW_U;        // N_CHUNK x ROW_U uints
    int*      idxl = (int*)(Bl1 + N_CHUNK * ROW_U);

    const int tid   = threadIdx.x;
    const int wave  = tid >> 5;
    const int lane  = tid & 31;
    const int l15   = lane & 15;
    const bool hi   = lane >= 16;
    const int mbase = blockIdx.x * M_BLK;
    const int wm    = wave * 16;                  // wave's 16-row slice

    const unsigned bl_off[2] = { (unsigned)(uintptr_t)(void*)Bl0,
                                 (unsigned)(uintptr_t)(void*)Bl1 };

    // ---- Prefetch codebook chunk 0 (async DMA to LDS, overlaps phase 1) ----
    {
        const char* g = (const char*)w_bf16;
        #pragma unroll
        for (int j = 0; j < B128_PER_THREAD; ++j) {
            const unsigned boff = (unsigned)(tid + j * 256) * 16u;
            async_load_b128(bl_off[0] + boff, g + boff);
        }
    }

    // ---- Phase 1: load + l2-normalize 16 x-rows per wave into LDS (bf16) ----
    for (int i = 0; i < 16; ++i) {
        const int lr = wm + i;
        const f4* xr = (const f4*)(x + (size_t)(mbase + lr) * CDIM);
        f4 p = __builtin_nontemporal_load(xr + lane * 2 + 0);
        f4 q = __builtin_nontemporal_load(xr + lane * 2 + 1);
        float s = p.x*p.x + p.y*p.y + p.z*p.z + p.w*p.w
                + q.x*q.x + q.y*q.y + q.z*q.z + q.w*q.w;
        float sc = row_scale(s);
        uint4 pb;
        pb.x = pk_bf16(p.x * sc, p.y * sc);
        pb.y = pk_bf16(p.z * sc, p.w * sc);
        pb.z = pk_bf16(q.x * sc, q.y * sc);
        pb.w = pk_bf16(q.z * sc, q.w * sc);
        *(uint4*)(Al + lr * ROW_U + lane * 4) = pb;
    }
    __syncthreads();

    // ---- Load this wave's A fragments (16 rows x K=256 -> 8 frags, 64 VGPRs) ----
    // A k order used here (same as B): lanes 0-15 hold K 0..15, lanes 16-31 K 16..31
    FragAB afrag[8];
    {
        const unsigned* arow = Al + (wm + l15) * ROW_U;
        #pragma unroll
        for (int kf = 0; kf < 8; ++kf) {
            const int off = kf * 16 + (hi ? 8 : 0);
            afrag[kf].q[0] = *(const uint4*)(arow + off);
            afrag[kf].q[1] = *(const uint4*)(arow + off + 4);
        }
    }

    float bestv[8];
    int   besti[8];
    #pragma unroll
    for (int r = 0; r < 8; ++r) { bestv[r] = -3.4e38f; besti[r] = 0; }

    // C/D layout: VGPR r, lanes 0-15 -> row r, col lane; lanes 16-31 -> row r+8
    const int m_row0 = mbase + wm + (hi ? 8 : 0);

    // ---- Phase 2: double-buffered async chunks + dual-chain pipelined WMMA ----
    for (int nc = 0; nc < N_CHUNKS; ++nc) {
        wait_async0();           // chunk nc resident in buf[nc&1]
        __syncthreads();

        if (nc + 1 < N_CHUNKS) { // prefetch next chunk into the other buffer
            const char* g = (const char*)(w_bf16 + (size_t)(nc + 1) * N_CHUNK * ROW_U);
            #pragma unroll
            for (int j = 0; j < B128_PER_THREAD; ++j) {
                const unsigned boff = (unsigned)(tid + j * 256) * 16u;
                async_load_b128(bl_off[(nc + 1) & 1] + boff, g + boff);
            }
        }

        const unsigned* cur = (nc & 1) ? Bl1 : Bl0;

        #pragma unroll
        for (int np = 0; np < N_CHUNK / 32; ++np) {
            v8f c0 = {}, c1 = {};
            // B k order used here (same as A): lanes 0-15 col=lane K 0..15, lanes 16-31 K 16..31
            const unsigned* brow0 = cur + ((np * 2 + 0) * 16 + l15) * ROW_U + (hi ? 8 : 0);
            const unsigned* brow1 = cur + ((np * 2 + 1) * 16 + l15) * ROW_U + (hi ? 8 : 0);

            // software-pipelined B fragments: issue kf+1 loads before kf WMMAs
            FragAB b0c, b1c, b0n, b1n;
            b0c.q[0] = *(const uint4*)(brow0 + 0);
            b0c.q[1] = *(const uint4*)(brow0 + 4);
            b1c.q[0] = *(const uint4*)(brow1 + 0);
            b1c.q[1] = *(const uint4*)(brow1 + 4);
            #pragma unroll
            for (int kf = 0; kf < 8; ++kf) {
                if (kf < 7) {
                    b0n.q[0] = *(const uint4*)(brow0 + (kf + 1) * 16);
                    b0n.q[1] = *(const uint4*)(brow0 + (kf + 1) * 16 + 4);
                    b1n.q[0] = *(const uint4*)(brow1 + (kf + 1) * 16);
                    b1n.q[1] = *(const uint4*)(brow1 + (kf + 1) * 16 + 4);
                }
                c0 = __builtin_amdgcn_wmma_f32_16x16x32_bf16(
                         false, afrag[kf].v, false, b0c.v, (short)0, c0, false, false);
                c1 = __builtin_amdgcn_wmma_f32_16x16x32_bf16(
                         false, afrag[kf].v, false, b1c.v, (short)0, c1, false, false);
                if (kf < 7) { b0c = b0n; b1c = b1n; }   // renamed by full unroll
            }

            const int col0 = nc * N_CHUNK + np * 32 + l15;
            float* drow = dist_out + (size_t)m_row0 * EDIM + col0;
            #pragma unroll
            for (int r = 0; r < 8; ++r) {
                const float v0 = c0[r], v1 = c1[r];
                __builtin_nontemporal_store(v0, drow + (size_t)r * EDIM);
                __builtin_nontemporal_store(v1, drow + (size_t)r * EDIM + 16);
                if (v0 > bestv[r]) { bestv[r] = v0; besti[r] = col0; }
                if (v1 > bestv[r]) { bestv[r] = v1; besti[r] = col0 + 16; }
            }
        }
    }

    // ---- Phase 3: argmax reduce across the 16 lanes holding one row group ----
    #pragma unroll
    for (int mask = 8; mask >= 1; mask >>= 1) {
        #pragma unroll
        for (int r = 0; r < 8; ++r) {
            const float ov = __shfl_xor(bestv[r], mask);
            const int   oi = __shfl_xor(besti[r], mask);
            if (ov > bestv[r] || (ov == bestv[r] && oi < besti[r])) {
                bestv[r] = ov; besti[r] = oi;
            }
        }
    }
    if (l15 == 0) {   // lane 0 -> rows wm..wm+7, lane 16 -> rows wm+8..wm+15
        #pragma unroll
        for (int r = 0; r < 8; ++r) {
            const int row_l = wm + (hi ? 8 : 0) + r;
            const int e = besti[r];
            idxl[row_l] = e;
            __builtin_nontemporal_store((float)e, embed_out + mbase + row_l);
            flags[e] = 1u;            // benign race: all writers store 1
        }
    }
    __syncthreads();

    // ---- Phase 4: gather quantized = w_norm[embed_idx] (full f32 precision) ----
    for (int i = 0; i < 16; ++i) {
        const int lr = wm + i;
        const int e  = idxl[lr];
        const f4* src = (const f4*)(w_norm + (size_t)e * CDIM);
        f4* dst = (f4*)(quant_out + (size_t)(mbase + lr) * CDIM);
        f4 a = src[lane * 2 + 0];
        f4 b = src[lane * 2 + 1];
        __builtin_nontemporal_store(a, dst + lane * 2 + 0);
        __builtin_nontemporal_store(b, dst + lane * 2 + 1);
    }
}

// ---------------------------------------------------------------------------
// Kernel 3: code_usage = 100 * (#unused codes) / E
// ---------------------------------------------------------------------------
__global__ void vq_usage(const unsigned* __restrict__ flags,
                         float* __restrict__ usage_out) {
    __shared__ int zc;
    if (threadIdx.x == 0) zc = 0;
    __syncthreads();
    int c = 0;
    for (int i = threadIdx.x; i < EDIM; i += 256) c += (flags[i] == 0u) ? 1 : 0;
    atomicAdd(&zc, c);
    __syncthreads();
    if (threadIdx.x == 0) *usage_out = 100.0f * ((float)zc / (float)EDIM);
}

// ---------------------------------------------------------------------------
extern "C" void kernel_launch(void* const* d_in, const int* in_sizes, int n_in,
                              void* d_out, int out_size, void* d_ws, size_t ws_size,
                              hipStream_t stream) {
    const float* x = (const float*)d_in[0];      // (8,16384,256) f32
    const float* w = (const float*)d_in[1];      // (1024,256) f32

    float* out = (float*)d_out;
    const size_t QUANT = (size_t)M_TOTAL * CDIM;     // 33,554,432
    const size_t DIST  = (size_t)M_TOTAL * EDIM;     // 134,217,728
    float* quant_out = out;
    float* dist_out  = out + QUANT;
    float* embed_out = out + QUANT + DIST;
    float* usage_out = out + QUANT + DIST + M_TOTAL;

    char* ws = (char*)d_ws;
    float*    w_norm = (float*)ws;                                   // 1 MB
    unsigned* w_bf16 = (unsigned*)(ws + (size_t)EDIM * CDIM * 4);    // 512 KB
    unsigned* flags  = (unsigned*)(ws + (size_t)EDIM * CDIM * 6);    // 4 KB

    vq_prep_weights<<<EDIM / 8, 256, 0, stream>>>(w, w_norm, w_bf16, flags);

    const size_t smem_bytes =
        (size_t)(M_BLK * ROW_U + 2 * N_CHUNK * ROW_U) * 4 + M_BLK * 4; // 131,584 B
    vq_main<<<M_TOTAL / M_BLK, 256, smem_bytes, stream>>>(
        x, w_bf16, w_norm, flags, quant_out, dist_out, embed_out);

    vq_usage<<<1, 256, 0, stream>>>(flags, usage_out);
}
